// RNN_50397146251808
// MI455X (gfx1250) — matrix operand order repaired and checked by reading, not compiled, not run on hardware
//
#include <hip/hip_runtime.h>
#include <hip/hip_bf16.h>
#include <stdint.h>

// ---------------------------------------------------------------------------
// Fused 2-layer tanh-RNN + FC for MI455X (gfx1250, wave32, WMMA).
//   B=256, T=1024, H=128, V=96
//
//  * embproj[v][o] = emb@W_ih0^T + biases  -> layer-0 input GEMM becomes a
//    96x128 lookup table (indices are vocab ids).
//  * Each workgroup owns a 16-row batch slice for the whole sequence;
//    8 waves x (16x16 N-tile); weights staged to LDS once, B-fragments
//    hoisted into registers for the entire time loop.
//  * Branchless tanh via v_exp_f32/v_rcp_f32 (critical-path friendly).
// ---------------------------------------------------------------------------

typedef __attribute__((ext_vector_type(16))) _Float16 v16h;
typedef __attribute__((ext_vector_type(8)))  float    v8f;

#define Bsz 256
#define Tsz 1024
#define Vsz 96
#define Hsz 128

// LDS layout (bytes)
#define SM_WHH0   0                       // 128*128 f16 = 32768
#define SM_W1     (SM_WHH0 + 128*128*2)   // 128*256 f16 = 65536
#define SM_FC     (SM_W1   + 128*256*2)   // 96*128 f16  = 24576
#define SM_EMB    (SM_FC   + 96*128*2)    // 96*128 f32  = 49152
#define SM_H0     (SM_EMB  + 96*128*4)    // 2 * 16*128 f16 = 8192
#define SM_H1     (SM_H0   + 2*16*128*2)  // 8192
#define SM_XID    (SM_H1   + 2*16*128*2)  // 2*16 int = 128
#define SM_TOTAL  (SM_XID  + 2*16*4)

// ---- branchless tanh: 1 - 2/(exp2(2*log2e*x)+1); saturates at +-1 ---------
__device__ __forceinline__ float fast_tanh(float x) {
    const float e = __builtin_amdgcn_exp2f(x * 2.8853900817779268f); // 2*log2(e)
    const float r = __builtin_amdgcn_rcpf(e + 1.0f);
    return __builtin_fmaf(-2.0f, r, 1.0f);
}

// ---- WMMA fragment loads (wave32 layouts per CDNA5 ISA 7.12.2) -------------

// A: 16xK tile, element (m,k). lane l: m = l&15, half-wave hi = l>>4.
// VGPR v(0..3): K = kb + 2v + {0,1} + hi*8 ; VGPR v(4..7): +16.
__device__ __forceinline__ v16h load_a_frag(const _Float16* __restrict__ base,
                                            int lane, int stride, int kb) {
    const int m  = lane & 15;
    const int hi = (lane >> 4) & 1;
    const _Float16* row = base + m * stride + kb + hi * 8;
    v16h a;
    uint32_t* ap = (uint32_t*)&a;
#pragma unroll
    for (int v = 0; v < 4; ++v) ap[v]     = *(const uint32_t*)(row + 2 * v);
#pragma unroll
    for (int v = 0; v < 4; ++v) ap[4 + v] = *(const uint32_t*)(row + 16 + 2 * v);
    return a;
}

// B: KxN tile stored as Wt[n][k] (row-major, row stride = `stride`).
// lane l: n = ntile*16 + (l&15). VGPR v(0..3): K = kb + 2v + {0,1} + hi*8 ; VGPR v(4..7): +16.
__device__ __forceinline__ v16h load_b_frag(const _Float16* __restrict__ wt,
                                            int lane, int stride, int kb, int ntile) {
    const int n  = ntile * 16 + (lane & 15);
    const int hi = (lane >> 4) & 1;
    const _Float16* col = wt + n * stride + kb + hi * 8;
    v16h b;
    uint32_t* bp = (uint32_t*)&b;
#pragma unroll
    for (int v = 0; v < 8; ++v) bp[v] = *(const uint32_t*)(col + (v >> 2) * 16 + 2 * (v & 3));
    return b;
}

#define WMMA_F16(a, b, c) \
    __builtin_amdgcn_wmma_f32_16x16x32_f16(false, (a), false, (b), (short)0, (c), false, false)

// ---------------------------------------------------------------------------
// Kernel 1: embproj[v][o] = emb[v,:].W_ih0[o,:] + b_ih0[o] + b_hh0[o]
// ---------------------------------------------------------------------------
__global__ void embproj_kernel(const float* __restrict__ emb,
                               const float* __restrict__ Wih0,
                               const float* __restrict__ bih0,
                               const float* __restrict__ bhh0,
                               float* __restrict__ ep) {
    const int v = blockIdx.x;
    const int o = threadIdx.x;
    float s = bih0[o] + bhh0[o];
#pragma unroll 8
    for (int h = 0; h < Hsz; ++h) s += emb[v * Hsz + h] * Wih0[o * Hsz + h];
    ep[v * Hsz + o] = s;
}

// ---------------------------------------------------------------------------
// Kernel 2: fused recurrence. grid = B/16 = 16 workgroups, 256 threads each.
// ---------------------------------------------------------------------------
__global__ void __launch_bounds__(256)
rnn_fused_kernel(const int*   __restrict__ x,
                 const float* __restrict__ Whh0,
                 const float* __restrict__ Wih1,
                 const float* __restrict__ Whh1,
                 const float* __restrict__ bih1,
                 const float* __restrict__ bhh1,
                 const float* __restrict__ fcW,
                 const float* __restrict__ fcb,
                 const float* __restrict__ embproj,
                 float* __restrict__ out) {
    extern __shared__ char smem[];
    _Float16* sWhh0 = (_Float16*)(smem + SM_WHH0);  // [n][k]
    _Float16* sW1   = (_Float16*)(smem + SM_W1);    // [n][k0..255] = [Wih1 | Whh1]
    _Float16* sFc   = (_Float16*)(smem + SM_FC);    // [n][k]
    float*    sEmb  = (float*)   (smem + SM_EMB);   // [v][o]
    _Float16* sH0   = (_Float16*)(smem + SM_H0);    // ping-pong 16x128
    _Float16* sH1   = (_Float16*)(smem + SM_H1);    // ping-pong 16x128
    int*      sXid  = (int*)     (smem + SM_XID);   // ping-pong 16 indices

    const int tid  = threadIdx.x;
    const int wave = tid >> 5;
    const int lane = tid & 31;
    const int ncol = lane & 15;
    const int m8   = (lane >> 4) ? 8 : 0;
    const int b0   = blockIdx.x * 16;
    const int ng   = wave * 16 + ncol;

    // ---- stage weights into LDS (one-time) ----
    for (int i = tid; i < 128 * 128; i += 256) sWhh0[i] = (_Float16)Whh0[i];
    for (int i = tid; i < 128 * 256; i += 256) {
        const int n = i >> 8, k = i & 255;
        const float w = (k < 128) ? Wih1[n * 128 + k] : Whh1[n * 128 + (k - 128)];
        sW1[i] = (_Float16)w;
    }
    for (int i = tid; i < 96 * 128; i += 256) sFc[i]  = (_Float16)fcW[i];
    for (int i = tid; i < 96 * 128; i += 256) sEmb[i] = embproj[i];
    for (int i = tid; i < 16 * 128; i += 256) { sH0[i] = (_Float16)0.f; sH1[i] = (_Float16)0.f; }
    if (tid < 16) sXid[tid] = x[(b0 + tid) * Tsz + 0];

    const int   fcTile = (wave < 6) ? wave : 0;          // clamp: keep EXEC uniform
    const float bias1  = bih1[ng] + bhh1[ng];
    const float biasFC = (wave < 6) ? fcb[ng] : 0.f;

    __syncthreads();

    // ---- hoist all loop-invariant B (weight) fragments into registers ----
    v16h bW0[4], bW1a[4], bW1b[4], bFc[4];
#pragma unroll
    for (int i = 0; i < 4; ++i) {
        bW0[i]  = load_b_frag(sWhh0, lane, 128, 32 * i, wave);
        bW1a[i] = load_b_frag(sW1,   lane, 256, 32 * i, wave);
        bW1b[i] = load_b_frag(sW1,   lane, 256, 128 + 32 * i, wave);
        bFc[i]  = load_b_frag(sFc,   lane, 128, 32 * i, fcTile);
    }

    for (int t = 0; t < Tsz; ++t) {
        const int cur = t & 1, nxt = cur ^ 1;
        const _Float16* h0c = sH0 + cur * 2048;
        _Float16*       h0n = sH0 + nxt * 2048;
        const _Float16* h1c = sH1 + cur * 2048;
        _Float16*       h1n = sH1 + nxt * 2048;
        const int*      xid = sXid + cur * 16;

        // ---- layer 0: h0' = tanh(embproj[x_t] + h0 @ Whh0^T) ----
        v8f c0;
#pragma unroll
        for (int e = 0; e < 8; ++e) c0[e] = 0.f;
#pragma unroll
        for (int i = 0; i < 4; ++i) {
            v16h a = load_a_frag(h0c, lane, 128, 32 * i);
            c0 = WMMA_F16(a, bW0[i], c0);
        }
#pragma unroll
        for (int e = 0; e < 8; ++e) {
            const int m = e + m8;
            h0n[m * 128 + ng] = (_Float16)fast_tanh(c0[e] + sEmb[xid[m] * 128 + ng]);
        }
        __syncthreads();

        // ---- layer 1: h1' = tanh(b1 + [h0', h1] @ [Wih1;Whh1]^T) ----
        v8f c1;
#pragma unroll
        for (int e = 0; e < 8; ++e) c1[e] = bias1;
#pragma unroll
        for (int i = 0; i < 4; ++i) {
            v16h a = load_a_frag(h0n, lane, 128, 32 * i);
            c1 = WMMA_F16(a, bW1a[i], c1);
        }
#pragma unroll
        for (int i = 0; i < 4; ++i) {
            v16h a = load_a_frag(h1c, lane, 128, 32 * i);
            c1 = WMMA_F16(a, bW1b[i], c1);
        }
#pragma unroll
        for (int e = 0; e < 8; ++e) {
            const int m = e + m8;
            h1n[m * 128 + ng] = (_Float16)fast_tanh(c1[e]);
        }
        __syncthreads();

        // ---- FC: out_t = h1' @ fcW^T + fc_b  (waves 0-5 cover V=96) ----
        if (wave < 6) {
            v8f cf;
#pragma unroll
            for (int e = 0; e < 8; ++e) cf[e] = biasFC;
#pragma unroll
            for (int i = 0; i < 4; ++i) {
                v16h a = load_a_frag(h1n, lane, 128, 32 * i);
                cf = WMMA_F16(a, bFc[i], cf);
            }
#pragma unroll
            for (int e = 0; e < 8; ++e) {
                const int m = e + m8;
                out[((size_t)(b0 + m) * Tsz + t) * Vsz + ng] = cf[e];
            }
        } else if (wave == 7 && lane < 16 && t + 1 < Tsz) {
            sXid[nxt * 16 + lane] = x[(b0 + lane) * Tsz + (t + 1)];
        }
        __syncthreads();
    }

    // ---- final hidden states: buffer written at t = T-1 is ((T-1)&1)^1 ----
    const int fin = ((Tsz - 1) & 1) ^ 1;
    const _Float16* hf0 = sH0 + fin * 2048;
    const _Float16* hf1 = sH1 + fin * 2048;
    float* hid = out + (size_t)Bsz * Tsz * Vsz;
    for (int i = tid; i < 16 * 128; i += 256) {
        const int m = i >> 7, k = i & 127;
        hid[(size_t)0 * Bsz * Hsz + (b0 + m) * Hsz + k] = (float)hf0[m * 128 + k];
        hid[(size_t)1 * Bsz * Hsz + (b0 + m) * Hsz + k] = (float)hf1[m * 128 + k];
    }
}

// ---------------------------------------------------------------------------
extern "C" void kernel_launch(void* const* d_in, const int* in_sizes, int n_in,
                              void* d_out, int out_size, void* d_ws, size_t ws_size,
                              hipStream_t stream) {
    const int*   x     = (const int*)  d_in[0];
    const float* emb   = (const float*)d_in[1];
    const float* Wih0  = (const float*)d_in[2];
    const float* Whh0  = (const float*)d_in[3];
    const float* bih0  = (const float*)d_in[4];
    const float* bhh0  = (const float*)d_in[5];
    const float* Wih1  = (const float*)d_in[6];
    const float* Whh1  = (const float*)d_in[7];
    const float* bih1  = (const float*)d_in[8];
    const float* bhh1  = (const float*)d_in[9];
    const float* fcW   = (const float*)d_in[10];
    const float* fcb   = (const float*)d_in[11];
    float*       out   = (float*)d_out;
    float*       ep    = (float*)d_ws;      // 96*128 f32 = 48 KB scratch

    (void)in_sizes; (void)n_in; (void)out_size; (void)ws_size;

    static_assert(SM_TOTAL <= 320 * 1024, "LDS budget");
    hipFuncSetAttribute((const void*)rnn_fused_kernel,
                        hipFuncAttributeMaxDynamicSharedMemorySize, SM_TOTAL);

    embproj_kernel<<<dim3(Vsz), dim3(Hsz), 0, stream>>>(emb, Wih0, bih0, bhh0, ep);

    rnn_fused_kernel<<<dim3(Bsz / 16), dim3(256), SM_TOTAL, stream>>>(
        x, Whh0, Wih1, Whh1, bih1, bhh1, fcW, fcb, ep, out);
}
